// HMP_MACELayer_77017353552158
// MI455X (gfx1250) — matrix operand order repaired and checked by reading, not compiled, not run on hardware
//
#include <hip/hip_runtime.h>
#include <hip/hip_bf16.h>

typedef __bf16 bf16;
typedef __attribute__((ext_vector_type(16))) __bf16 v16bf;
typedef __attribute__((ext_vector_type(8)))  float  v8f;
typedef __attribute__((ext_vector_type(8)))  unsigned int v8u;

#define DEV_INLINE __device__ __forceinline__

constexpr int   NN    = 4096;
constexpr int   EE    = 131072;
constexpr int   DD    = 128;
constexpr int   DD2   = 256;
constexpr int   SS    = 128;
constexpr int   RR    = 8;
constexpr int   LL    = 16;
constexpr int   HIDN  = 64;
constexpr int   TK    = 16;
constexpr int   VV    = NN * TK;       // 65536 virtual edges
constexpr int   KA    = 288;           // 2S + R = 264, padded to 9*32
constexpr float NEGF  = -1.0e9f;
constexpr float GAM   = 4.0f;

// ---------------- scalar helpers ----------------
DEV_INLINE bf16 f2bf(float f) {
  unsigned u = __float_as_uint(f);
  unsigned r = u + 0x7FFFu + ((u >> 16) & 1u);   // round-to-nearest-even
  unsigned short h = (unsigned short)(r >> 16);
  return __builtin_bit_cast(bf16, h);
}
DEV_INLINE unsigned fenc(float f) {               // monotonic float->uint
  unsigned u = __float_as_uint(f);
  return (u & 0x80000000u) ? ~u : (u | 0x80000000u);
}
DEV_INLINE float fdec(unsigned e) {
  unsigned u = (e & 0x80000000u) ? (e ^ 0x80000000u) : ~e;
  return __uint_as_float(u);
}
DEV_INLINE void compute_sh(float dx, float dy, float dz, float len, float* sh) {
  float inv = 1.f / (len + 1e-9f);
  float x = dx * inv, y = dy * inv, z = dz * inv;
  sh[0] = 1.f;   sh[1] = x;     sh[2] = y;     sh[3] = z;
  sh[4] = x*x;   sh[5] = y*y;   sh[6] = z*z;
  sh[7] = x*y;   sh[8] = x*z;   sh[9] = y*z;
  sh[10] = x*x*x; sh[11] = y*y*y; sh[12] = z*z*z;
  sh[13] = x*x*y; sh[14] = y*y*z; sh[15] = z*z*x;
}

// ---------------- WMMA fragment loaders (16-bit, contiguous k = 16*half + s order) ------
// A 16x32 (MxK): lanes 0-15 -> M=0..15 with K 0..15; lanes 16-31 same M, K 16..31
DEV_INLINE v16bf load_fragA_rm(const bf16* A, int lda, int lane) {
  int m = lane & 15, hf = lane >> 4;
  const bf16* row = A + (size_t)m * lda;
  v8u r;
#pragma unroll
  for (int j = 0; j < 8; ++j) {
    int kb = 16 * hf + 2 * j;
    r[j] = *(const unsigned int*)(row + kb);      // 2 contiguous bf16
  }
  return __builtin_bit_cast(v16bf, r);
}
// B 32x16 (KxN) from row-major KxN source: lanes 0-15 -> K=0..15, lanes 16-31 -> K=16..31, N=lane&15
DEV_INLINE v16bf load_fragB_nt(const bf16* B, int ldb, int lane) {
  int n = lane & 15, hf = lane >> 4;
  v8u r;
#pragma unroll
  for (int j = 0; j < 8; ++j) {
    int k = 16 * hf + 2 * j;
    unsigned lo = __builtin_bit_cast(unsigned short, B[(size_t)k * ldb + n]);
    unsigned hi = __builtin_bit_cast(unsigned short, B[(size_t)(k + 1) * ldb + n]);
    r[j] = lo | (hi << 16);
  }
  return __builtin_bit_cast(v16bf, r);
}
// B 32x16 gathered from row-major NxK source (i.e. B = src^T): contiguous K -> 32-bit loads
DEV_INLINE v16bf load_fragB_tr(const bf16* Bt, int ldbt, int lane) {
  int n = lane & 15, hf = lane >> 4;
  const bf16* row = Bt + (size_t)n * ldbt;
  v8u r;
#pragma unroll
  for (int j = 0; j < 8; ++j) {
    int k = 16 * hf + 2 * j;
    r[j] = *(const unsigned int*)(row + k);
  }
  return __builtin_bit_cast(v16bf, r);
}

// ------- generic bf16 WMMA GEMM, 16x64 per wave: C = scale*(A@B) [+ pad] --
// grid = (Ncols/64, M/16), block = 32. A fragment reused across 4 B tiles.
__global__ void gemm_bf16(const bf16* __restrict__ A, int lda,
                          const bf16* __restrict__ B, int ldb, int transB,
                          float* __restrict__ C, int ldc, int K,
                          const float* __restrict__ Cadd, int addLd, int addCols,
                          float scale)
{
  int lane = threadIdx.x;
  int tn = blockIdx.x, tm = blockIdx.y;
  const bf16* Abase = A + (size_t)(tm * 16) * lda;
  v8f acc[4];
#pragma unroll
  for (int t = 0; t < 4; ++t) acc[t] = v8f{0.f, 0.f, 0.f, 0.f, 0.f, 0.f, 0.f, 0.f};

  if (transB) {
    for (int k0 = 0; k0 < K; k0 += 32) {
      v16bf a = load_fragA_rm(Abase + k0, lda, lane);
#pragma unroll
      for (int t = 0; t < 4; ++t) {
        v16bf b = load_fragB_tr(B + (size_t)(tn * 64 + t * 16) * ldb + k0, ldb, lane);
        acc[t] = __builtin_amdgcn_wmma_f32_16x16x32_bf16(false, a, false, b, (short)0, acc[t], false, false);
      }
    }
  } else {
    for (int k0 = 0; k0 < K; k0 += 32) {
      v16bf a = load_fragA_rm(Abase + k0, lda, lane);
#pragma unroll
      for (int t = 0; t < 4; ++t) {
        v16bf b = load_fragB_nt(B + (size_t)k0 * ldb + tn * 64 + t * 16, ldb, lane);
        acc[t] = __builtin_amdgcn_wmma_f32_16x16x32_bf16(false, a, false, b, (short)0, acc[t], false, false);
      }
    }
  }
  int n = lane & 15, hf = lane >> 4;
#pragma unroll
  for (int t = 0; t < 4; ++t) {
    int col = tn * 64 + t * 16 + n;
#pragma unroll
    for (int r = 0; r < 8; ++r) {
      int row = tm * 16 + r + 8 * hf;
      float v = acc[t][r] * scale;
      if (Cadd) v += (col < addCols) ? Cadd[(size_t)row * addLd + col] : 0.f;
      C[(size_t)row * ldc + col] = v;
    }
  }
}

// ---------------- utility kernels ----------------
__global__ void zero_f32(float* p, int n) {
  int i = blockIdx.x * 256 + threadIdx.x;
  if (i < n) p[i] = 0.f;
}
__global__ void cvt_f32_bf16(const float* __restrict__ s, bf16* __restrict__ d, int n) {
  int i = blockIdx.x * 256 + threadIdx.x;
  if (i < n) d[i] = f2bf(s[i]);
}
__global__ void pad_wa1(const float* __restrict__ W_a1, bf16* __restrict__ dst) {
  int idx = blockIdx.x * 256 + threadIdx.x;
  if (idx >= KA * SS) return;
  int k = idx >> 7, n = idx & 127;
  dst[idx] = (k < 2 * SS + RR) ? f2bf(W_a1[(size_t)k * SS + n])
                               : __builtin_bit_cast(bf16, (unsigned short)0);
}

// ---------------- stage 1: real-edge messages (gate = feats@W_loc_r) ------
__global__ void msgs1_kernel(const float* __restrict__ h,
                             const int* __restrict__ ei,
                             const float* __restrict__ esh,
                             const float* __restrict__ efe,
                             const float* __restrict__ W_loc_r,
                             const float* __restrict__ w_loc_sh,
                             float* __restrict__ acc1)
{
  __shared__ float sWr[RR * DD];
  __shared__ float sWsh[LL];
  int tid = threadIdx.x;
  for (int i = tid; i < RR * DD; i += 256) sWr[i] = W_loc_r[i];
  if (tid < LL) sWsh[tid] = w_loc_sh[tid];
  __syncthreads();
  int e = blockIdx.x * 8 + (tid >> 5);
  int lane = tid & 31;
  int srcn = ei[e], dstn = ei[EE + e];
  float s = 0.f;
#pragma unroll
  for (int l = 0; l < LL; ++l) s += esh[(size_t)e * LL + l] * sWsh[l];
  float ef[RR];
#pragma unroll
  for (int r = 0; r < RR; ++r) ef[r] = efe[(size_t)e * RR + r];
#pragma unroll
  for (int q = 0; q < DD / 32; ++q) {
    int d = lane + 32 * q;
    float gate = 0.f;
#pragma unroll
    for (int r = 0; r < RR; ++r) gate += ef[r] * sWr[r * DD + d];
    atomicAdd(&acc1[(size_t)dstn * DD + d], h[(size_t)srcn * DD + d] * gate * s);
  }
}

// ---------------- stage 4: gating MLP (S->HID->1, sigmoid) ----------------
__global__ void gate_mlp(const float* __restrict__ h_local,
                         const float* __restrict__ W1, const float* __restrict__ b1,
                         const float* __restrict__ W2, const float* __restrict__ b2,
                         float* __restrict__ m, float* __restrict__ mfv)
{
  __shared__ float red[HIDN];
  int i = blockIdx.x, tid = threadIdx.x;
  float acc = b1[tid];
  for (int s = 0; s < SS; ++s) acc += h_local[(size_t)i * DD2 + s] * W1[s * HIDN + tid];
  acc = fmaxf(acc, 0.f);
  red[tid] = acc * W2[tid];
  __syncthreads();
  for (int s = HIDN / 2; s > 0; s >>= 1) {
    if (tid < s) red[tid] += red[tid + s];
    __syncthreads();
  }
  if (tid == 0) {
    float z = red[0] + b2[0];
    float mm = 1.f / (1.f + expf(-z));
    m[i] = mm;
    mfv[i] = (mm > 0.5f) ? 1.f : 0.f;
  }
}

// ---------------- stage 7: masked row-wise top-16 -------------------------
__global__ void topk_kernel(const float* __restrict__ scoresP,
                            const float* __restrict__ mfv, int rowBase,
                            float* __restrict__ topv, int* __restrict__ topi)
{
  __shared__ float svals[NN];
  __shared__ float rv[256];
  __shared__ int   ri[256];
  int tid = threadIdx.x;
  int row = rowBase + blockIdx.x;
  float mi = mfv[row];
  for (int j = tid; j < NN; j += 256) {
    float v = scoresP[(size_t)blockIdx.x * NN + j];
    bool ok = (mi > 0.5f) && (mfv[j] > 0.5f) && (j != row);
    svals[j] = ok ? v : NEGF;
  }
  __syncthreads();
  for (int t = 0; t < TK; ++t) {
    float best = -3.0e38f; int bi = 0;
    for (int j = tid; j < NN; j += 256) {
      float v = svals[j];
      if (v > best) { best = v; bi = j; }
    }
    rv[tid] = best; ri[tid] = bi;
    __syncthreads();
    for (int s = 128; s > 0; s >>= 1) {
      if (tid < s) {
        if (rv[tid + s] > rv[tid] || (rv[tid + s] == rv[tid] && ri[tid + s] < ri[tid])) {
          rv[tid] = rv[tid + s]; ri[tid] = ri[tid + s];
        }
      }
      __syncthreads();
    }
    if (tid == 0) {
      topv[(size_t)row * TK + t] = rv[0];
      topi[(size_t)row * TK + t] = ri[0];
      svals[ri[0]] = -3.4e38f;
    }
    __syncthreads();
  }
}

// ---------------- stage 8a: virtual edge geometry -------------------------
__global__ void virt_setup(const float* __restrict__ pos,
                           const int* __restrict__ topi,
                           float* __restrict__ sh_v, float* __restrict__ rf_v)
{
  int v = blockIdx.x * 256 + threadIdx.x;
  if (v >= VV) return;
  int i = v >> 4, j = topi[v];
  float dx = pos[i * 3 + 0] - pos[j * 3 + 0];
  float dy = pos[i * 3 + 1] - pos[j * 3 + 1];
  float dz = pos[i * 3 + 2] - pos[j * 3 + 2];
  float len = sqrtf(dx * dx + dy * dy + dz * dz);
  float sh[LL];
  compute_sh(dx, dy, dz, len, sh);
#pragma unroll
  for (int l = 0; l < LL; ++l) sh_v[(size_t)v * LL + l] = sh[l];
#pragma unroll
  for (int r = 0; r < RR; ++r) {
    float t = len - (5.f / 7.f) * (float)r;
    rf_v[(size_t)v * RR + r] = expf(-GAM * t * t);
  }
}

// ---------------- stage 8b: fused WMMA attention MLP ----------------------
// one wave per 16 virtual edges: A = [h_s[src] | h_s[dst] | rf | 0pad] (16x288 bf16 in LDS)
__global__ void attn_mlp(const bf16* __restrict__ hloc_bf,
                         const int* __restrict__ topi,
                         const float* __restrict__ topv,
                         const float* __restrict__ rf_v,
                         const bf16* __restrict__ Wa1p,     // KA x SS
                         const float* __restrict__ b_a1,
                         const float* __restrict__ W_a2,
                         const float* __restrict__ b_a2,
                         float* __restrict__ sc)
{
  __shared__ bf16 tileA[16 * KA];
  int lane = threadIdx.x;
  int base = blockIdx.x * 16;
  for (int idx = lane; idx < 16 * KA; idx += 32) {
    int mrow = idx / KA, c = idx - mrow * KA;
    int e = base + mrow;
    bf16 val = __builtin_bit_cast(bf16, (unsigned short)0);
    if (c < SS) {
      val = hloc_bf[(size_t)(e >> 4) * DD2 + c];
    } else if (c < 2 * SS) {
      val = hloc_bf[(size_t)topi[e] * DD2 + (c - SS)];
    } else if (c < 2 * SS + RR) {
      val = f2bf(rf_v[(size_t)e * RR + (c - 2 * SS)]);
    }
    tileA[idx] = val;
  }
  __syncthreads();

  float part[8];
#pragma unroll
  for (int r = 0; r < 8; ++r) part[r] = 0.f;
  int ncol = lane & 15, hf = lane >> 4;
  for (int nc = 0; nc < SS / 16; ++nc) {
    v8f c = {0.f, 0.f, 0.f, 0.f, 0.f, 0.f, 0.f, 0.f};
#pragma unroll
    for (int kt = 0; kt < KA / 32; ++kt) {
      v16bf a = load_fragA_rm(&tileA[kt * 32], KA, lane);
      v16bf b = load_fragB_nt(Wa1p + (size_t)(kt * 32) * SS + nc * 16, SS, lane);
      c = __builtin_amdgcn_wmma_f32_16x16x32_bf16(false, a, false, b, (short)0, c, false, false);
    }
    float bb = b_a1[nc * 16 + ncol];
    float a2 = W_a2[nc * 16 + ncol];
#pragma unroll
    for (int r = 0; r < 8; ++r) {
      float hcol = fmaxf(c[r] + bb, 0.f);       // bias + relu
      part[r] += hcol * a2;                     // dot with W_a2 column weights
    }
  }
#pragma unroll
  for (int r = 0; r < 8; ++r) {                 // reduce across the 16 hidden-col lanes
    part[r] += __shfl_xor(part[r], 1, 32);
    part[r] += __shfl_xor(part[r], 2, 32);
    part[r] += __shfl_xor(part[r], 4, 32);
    part[r] += __shfl_xor(part[r], 8, 32);
  }
  if (ncol == 0) {
    float b2 = b_a2[0];
#pragma unroll
    for (int r = 0; r < 8; ++r) {
      int e = base + 8 * hf + r;
      bool valid = topv[e] > NEGF * 0.5f;
      sc[e] = valid ? (part[r] + b2) : NEGF;
    }
  }
}

// ---------------- stage 9: segment softmax over vdst ----------------------
__global__ void smax_init(unsigned* __restrict__ mx_u, float* __restrict__ den) {
  int i = blockIdx.x * 256 + threadIdx.x;
  if (i < NN) { mx_u[i] = fenc(NEGF); den[i] = 0.f; }
}
__global__ void smax_max(const float* __restrict__ sc, const int* __restrict__ topi,
                         unsigned* __restrict__ mx_u) {
  int v = blockIdx.x * 256 + threadIdx.x;
  if (v < VV) atomicMax(&mx_u[topi[v]], fenc(sc[v]));
}
__global__ void smax_ex(const float* __restrict__ sc, const float* __restrict__ topv,
                        const int* __restrict__ topi, const unsigned* __restrict__ mx_u,
                        float* __restrict__ exv, float* __restrict__ den) {
  int v = blockIdx.x * 256 + threadIdx.x;
  if (v >= VV) return;
  float ex = 0.f;
  if (topv[v] > NEGF * 0.5f) ex = expf(sc[v] - fdec(mx_u[topi[v]]));
  exv[v] = ex;
  if (ex != 0.f) atomicAdd(&den[topi[v]], ex);
}
__global__ void smax_wv(const float* __restrict__ exv, const int* __restrict__ topi,
                        const float* __restrict__ den, float* __restrict__ wv) {
  int v = blockIdx.x * 256 + threadIdx.x;
  if (v < VV) wv[v] = exv[v] / (den[topi[v]] + 1e-12f);
}

// ---------------- stage 10: second message passes -------------------------
__global__ void msgs2_real(const float* __restrict__ h_local,
                           const int* __restrict__ ei,
                           const float* __restrict__ pos,
                           const float* __restrict__ mfv,
                           const float* __restrict__ W_h_r,
                           const float* __restrict__ w_h_sh,
                           float* __restrict__ agg)
{
  __shared__ float sWr[RR * DD2];
  __shared__ float sWsh[LL];
  int tid = threadIdx.x;
  for (int i = tid; i < RR * DD2; i += 256) sWr[i] = W_h_r[i];
  if (tid < LL) sWsh[tid] = w_h_sh[tid];
  __syncthreads();
  int e = blockIdx.x * 8 + (tid >> 5);
  int lane = tid & 31;
  int srcn = ei[e], dstn = ei[EE + e];
  float pair = mfv[srcn] * mfv[dstn];
  if (pair == 0.f) return;
  float dx = pos[srcn * 3 + 0] - pos[dstn * 3 + 0];
  float dy = pos[srcn * 3 + 1] - pos[dstn * 3 + 1];
  float dz = pos[srcn * 3 + 2] - pos[dstn * 3 + 2];
  float len = sqrtf(dx * dx + dy * dy + dz * dz);
  float sh[LL];
  compute_sh(dx, dy, dz, len, sh);
  float s = 0.f;
#pragma unroll
  for (int l = 0; l < LL; ++l) s += sh[l] * sWsh[l];
  float rf[RR];
#pragma unroll
  for (int r = 0; r < RR; ++r) {
    float t = len - (5.f / 7.f) * (float)r;
    rf[r] = expf(-GAM * t * t) * pair;
  }
#pragma unroll
  for (int q = 0; q < DD2 / 32; ++q) {
    int d = lane + 32 * q;
    float gate = 0.f;
#pragma unroll
    for (int r = 0; r < RR; ++r) gate += rf[r] * sWr[r * DD2 + d];
    atomicAdd(&agg[(size_t)dstn * DD2 + d], h_local[(size_t)srcn * DD2 + d] * gate * s);
  }
}

__global__ void msgs2_virt(const float* __restrict__ h_local,
                           const int* __restrict__ topi,
                           const float* __restrict__ sh_v,
                           const float* __restrict__ rf_v,
                           const float* __restrict__ wv,
                           const float* __restrict__ W_h_r,
                           const float* __restrict__ w_h_sh,
                           float* __restrict__ agg)
{
  __shared__ float sWr[RR * DD2];
  __shared__ float sWsh[LL];
  int tid = threadIdx.x;
  for (int i = tid; i < RR * DD2; i += 256) sWr[i] = W_h_r[i];
  if (tid < LL) sWsh[tid] = w_h_sh[tid];
  __syncthreads();
  int v = blockIdx.x * 8 + (tid >> 5);
  int lane = tid & 31;
  float w = wv[v];
  if (w == 0.f) return;
  int srcn = v >> 4, dstn = topi[v];
  float s = 0.f;
#pragma unroll
  for (int l = 0; l < LL; ++l) s += sh_v[(size_t)v * LL + l] * sWsh[l];
  float rf[RR];
#pragma unroll
  for (int r = 0; r < RR; ++r) rf[r] = rf_v[(size_t)v * RR + r] * w;
#pragma unroll
  for (int q = 0; q < DD2 / 32; ++q) {
    int d = lane + 32 * q;
    float gate = 0.f;
#pragma unroll
    for (int r = 0; r < RR; ++r) gate += rf[r] * sWr[r * DD2 + d];
    atomicAdd(&agg[(size_t)dstn * DD2 + d], h_local[(size_t)srcn * DD2 + d] * gate * s);
  }
}

// ---------------- stage 12: final combine ---------------------------------
__global__ void final_combine(const float* __restrict__ h_local,
                              const float* __restrict__ hh,
                              const float* __restrict__ m,
                              const float* __restrict__ mfv,
                              float* __restrict__ out)
{
  int idx = blockIdx.x * 256 + threadIdx.x;
  if (idx >= NN * DD2) return;
  int i = idx >> 8;
  float mm = m[i];
  float hl = h_local[idx];
  float hhier = (hh[idx] + hl) * mfv[i];
  out[idx] = (1.f - mm) * hl + mm * hhier;
}

// ---------------- host side ----------------------------------------------
extern "C" void kernel_launch(void* const* d_in, const int* in_sizes, int n_in,
                              void* d_out, int out_size, void* d_ws, size_t ws_size,
                              hipStream_t stream) {
  (void)in_sizes; (void)n_in; (void)out_size; (void)ws_size;
  const float* h         = (const float*)d_in[0];
  const float* pos       = (const float*)d_in[1];
  const int*   ei        = (const int*)  d_in[2];
  const float* esh       = (const float*)d_in[3];
  const float* efe       = (const float*)d_in[4];
  const float* W_loc_r   = (const float*)d_in[6];
  const float* w_loc_sh  = (const float*)d_in[7];
  const float* W_loc_out = (const float*)d_in[8];
  const float* W_prod_loc= (const float*)d_in[9];
  const float* W_ms1     = (const float*)d_in[10];
  const float* b_ms1     = (const float*)d_in[11];
  const float* W_ms2     = (const float*)d_in[12];
  const float* b_ms2     = (const float*)d_in[13];
  const float* Wq        = (const float*)d_in[14];
  const float* Wk        = (const float*)d_in[15];
  const float* W_a1      = (const float*)d_in[16];
  const float* b_a1      = (const float*)d_in[17];
  const float* W_a2      = (const float*)d_in[18];
  const float* b_a2      = (const float*)d_in[19];
  const float* W_h_r     = (const float*)d_in[20];
  const float* w_h_sh    = (const float*)d_in[21];
  const float* W_h_out   = (const float*)d_in[22];
  const float* W_prod_h  = (const float*)d_in[23];
  float* out = (float*)d_out;

  // ---- carve workspace ----
  char* W = (char*)d_ws;
  size_t off = 0;
  auto alloc = [&](size_t bytes) -> void* {
    void* p = W + off;
    off += (bytes + 255) & ~(size_t)255;
    return p;
  };
  float* acc1    = (float*)alloc((size_t)NN * DD * 4);
  float* h_up    = (float*)alloc((size_t)NN * DD2 * 4);
  float* h_loc   = (float*)alloc((size_t)NN * DD2 * 4);
  float* qf      = (float*)alloc((size_t)NN * SS * 4);
  float* kf      = (float*)alloc((size_t)NN * SS * 4);
  float* mbuf    = (float*)alloc((size_t)NN * 4);
  float* mfv     = (float*)alloc((size_t)NN * 4);
  unsigned* mx_u = (unsigned*)alloc((size_t)NN * 4);
  float* den     = (float*)alloc((size_t)NN * 4);
  float* topv    = (float*)alloc((size_t)VV * 4);
  int*   topi    = (int*)  alloc((size_t)VV * 4);
  float* sh_v    = (float*)alloc((size_t)VV * LL * 4);
  float* rf_v    = (float*)alloc((size_t)VV * RR * 4);
  float* sc      = (float*)alloc((size_t)VV * 4);
  float* exv     = (float*)alloc((size_t)VV * 4);
  float* wv      = (float*)alloc((size_t)VV * 4);
  float* agg     = (float*)alloc((size_t)NN * DD2 * 4);
  float* hmup    = (float*)alloc((size_t)NN * DD2 * 4);
  float* hh      = (float*)alloc((size_t)NN * DD2 * 4);
  float* scoresP = (float*)alloc((size_t)512 * NN * 4);   // row panel
  bf16* acc1_bf  = (bf16*)alloc((size_t)NN * DD * 2);
  bf16* hup_bf   = (bf16*)alloc((size_t)NN * DD2 * 2);
  bf16* hloc_bf  = (bf16*)alloc((size_t)NN * DD2 * 2);
  bf16* q_bf     = (bf16*)alloc((size_t)NN * SS * 2);
  bf16* k_bf     = (bf16*)alloc((size_t)NN * SS * 2);
  bf16* agg_bf   = (bf16*)alloc((size_t)NN * DD2 * 2);
  bf16* hmup_bf  = (bf16*)alloc((size_t)NN * DD2 * 2);
  bf16* Wlo_bf   = (bf16*)alloc((size_t)DD * DD2 * 2);
  bf16* Wpl_bf   = (bf16*)alloc((size_t)DD2 * DD2 * 2);
  bf16* Wq_bf    = (bf16*)alloc((size_t)SS * SS * 2);
  bf16* Wk_bf    = (bf16*)alloc((size_t)SS * SS * 2);
  bf16* Wa1_bf   = (bf16*)alloc((size_t)KA * SS * 2);
  bf16* Who_bf   = (bf16*)alloc((size_t)DD2 * DD2 * 2);
  bf16* Wph_bf   = (bf16*)alloc((size_t)DD2 * DD2 * 2);

  auto cdiv = [](int a, int b) { return (a + b - 1) / b; };
  const float invsq = 1.f / (sqrtf((float)SS) * 1.0f /*LAMBDA*/);

  // weight conversion
  cvt_f32_bf16<<<cdiv(DD * DD2, 256), 256, 0, stream>>>(W_loc_out, Wlo_bf, DD * DD2);
  cvt_f32_bf16<<<cdiv(DD2 * DD2, 256), 256, 0, stream>>>(W_prod_loc, Wpl_bf, DD2 * DD2);
  cvt_f32_bf16<<<cdiv(SS * SS, 256), 256, 0, stream>>>(Wq, Wq_bf, SS * SS);
  cvt_f32_bf16<<<cdiv(SS * SS, 256), 256, 0, stream>>>(Wk, Wk_bf, SS * SS);
  cvt_f32_bf16<<<cdiv(DD2 * DD2, 256), 256, 0, stream>>>(W_h_out, Who_bf, DD2 * DD2);
  cvt_f32_bf16<<<cdiv(DD2 * DD2, 256), 256, 0, stream>>>(W_prod_h, Wph_bf, DD2 * DD2);
  pad_wa1<<<cdiv(KA * SS, 256), 256, 0, stream>>>(W_a1, Wa1_bf);

  // stage 1: first message pass
  zero_f32<<<cdiv(NN * DD, 256), 256, 0, stream>>>(acc1, NN * DD);
  msgs1_kernel<<<EE / 8, 256, 0, stream>>>(h, ei, esh, efe, W_loc_r, w_loc_sh, acc1);
  cvt_f32_bf16<<<cdiv(NN * DD, 256), 256, 0, stream>>>(acc1, acc1_bf, NN * DD);

  // stage 2: h_up = acc1 @ W_loc_out    (Ncols=256 -> grid.x = 4)
  gemm_bf16<<<dim3(DD2 / 64, NN / 16), 32, 0, stream>>>(acc1_bf, DD, Wlo_bf, DD2, 0,
      h_up, DD2, DD, nullptr, 0, 0, 1.f);
  cvt_f32_bf16<<<cdiv(NN * DD2, 256), 256, 0, stream>>>(h_up, hup_bf, NN * DD2);

  // stage 3: h_local = h_up @ W_prod_loc + pad(h)
  gemm_bf16<<<dim3(DD2 / 64, NN / 16), 32, 0, stream>>>(hup_bf, DD2, Wpl_bf, DD2, 0,
      h_loc, DD2, DD2, h, DD, DD, 1.f);
  cvt_f32_bf16<<<cdiv(NN * DD2, 256), 256, 0, stream>>>(h_loc, hloc_bf, NN * DD2);

  // stage 4: gating MLP -> m, mf
  gate_mlp<<<NN, HIDN, 0, stream>>>(h_loc, W_ms1, b_ms1, W_ms2, b_ms2, mbuf, mfv);

  // stage 5: q, k projections (h_scalar = first 128 cols of h_local)
  gemm_bf16<<<dim3(SS / 64, NN / 16), 32, 0, stream>>>(hloc_bf, DD2, Wq_bf, SS, 0,
      qf, SS, SS, nullptr, 0, 0, 1.f);
  gemm_bf16<<<dim3(SS / 64, NN / 16), 32, 0, stream>>>(hloc_bf, DD2, Wk_bf, SS, 0,
      kf, SS, SS, nullptr, 0, 0, 1.f);
  cvt_f32_bf16<<<cdiv(NN * SS, 256), 256, 0, stream>>>(qf, q_bf, NN * SS);
  cvt_f32_bf16<<<cdiv(NN * SS, 256), 256, 0, stream>>>(kf, k_bf, NN * SS);

  // stage 6+7: panelled scores GEMM (q @ k^T) + masked top-16
  for (int p = 0; p < NN / 512; ++p) {
    gemm_bf16<<<dim3(NN / 64, 512 / 16), 32, 0, stream>>>(q_bf + (size_t)p * 512 * SS, SS,
        k_bf, SS, 1, scoresP, NN, SS, nullptr, 0, 0, invsq);
    topk_kernel<<<512, 256, 0, stream>>>(scoresP, mfv, p * 512, topv, topi);
  }

  // stage 8: virtual edge geometry + fused WMMA attention MLP
  virt_setup<<<VV / 256, 256, 0, stream>>>(pos, topi, sh_v, rf_v);
  attn_mlp<<<VV / 16, 32, 0, stream>>>(hloc_bf, topi, topv, rf_v, Wa1_bf, b_a1, W_a2, b_a2, sc);

  // stage 9: segment softmax over vdst
  smax_init<<<cdiv(NN, 256), 256, 0, stream>>>(mx_u, den);
  smax_max<<<VV / 256, 256, 0, stream>>>(sc, topi, mx_u);
  smax_ex<<<VV / 256, 256, 0, stream>>>(sc, topv, topi, mx_u, exv, den);
  smax_wv<<<VV / 256, 256, 0, stream>>>(exv, topi, den, wv);

  // stage 10: second message passes into agg
  zero_f32<<<cdiv(NN * DD2, 256), 256, 0, stream>>>(agg, NN * DD2);
  msgs2_real<<<EE / 8, 256, 0, stream>>>(h_loc, ei, pos, mfv, W_h_r, w_h_sh, agg);
  msgs2_virt<<<VV / 8, 256, 0, stream>>>(h_loc, topi, sh_v, rf_v, wv, W_h_r, w_h_sh, agg);
  cvt_f32_bf16<<<cdiv(NN * DD2, 256), 256, 0, stream>>>(agg, agg_bf, NN * DD2);

  // stage 11: h_m_up = agg @ W_h_out ; hh = h_m_up @ W_prod_h
  gemm_bf16<<<dim3(DD2 / 64, NN / 16), 32, 0, stream>>>(agg_bf, DD2, Who_bf, DD2, 0,
      hmup, DD2, DD2, nullptr, 0, 0, 1.f);
  cvt_f32_bf16<<<cdiv(NN * DD2, 256), 256, 0, stream>>>(hmup, hmup_bf, NN * DD2);
  gemm_bf16<<<dim3(DD2 / 64, NN / 16), 32, 0, stream>>>(hmup_bf, DD2, Wph_bf, DD2, 0,
      hh, DD2, DD2, nullptr, 0, 0, 1.f);

  // stage 12: final combine
  final_combine<<<cdiv(NN * DD2, 256), 256, 0, stream>>>(h_loc, hh, mbuf, mfv, out);
}
